// MMD_46866683134277
// MI455X (gfx1250) — matrix operand order repaired and checked by reading, not compiled, not run on hardware
//
#include <hip/hip_runtime.h>
#include <math.h>

typedef _Float16 half_t;
typedef __attribute__((ext_vector_type(16))) _Float16 v16h;
typedef __attribute__((ext_vector_type(8)))  _Float16 v8h;
typedef __attribute__((ext_vector_type(8)))  float    v8f;

#define NX    8192
#define DIM   256
#define ZROWS 16384
#define TILE  128
#define NT    (ZROWS / TILE)   // 128 tile rows/cols

// ---------------------------------------------------------------------------
// Kernel 1: split Z=[x;y] rows into f16 hi/lo halves and compute row norms.
// One 256-thread block per row (DIM == 256).
// ---------------------------------------------------------------------------
__global__ __launch_bounds__(256) void mmd_prep(
    const float* __restrict__ x, const float* __restrict__ y,
    half_t* __restrict__ zhi, half_t* __restrict__ zlo,
    float* __restrict__ z2)
{
  const int row = blockIdx.x;
  const int d   = threadIdx.x;
  const float v = (row < NX) ? x[(size_t)row * DIM + d]
                             : y[(size_t)(row - NX) * DIM + d];
  const half_t h = (half_t)v;
  const half_t l = (half_t)(v - (float)h);
  zhi[(size_t)row * DIM + d] = h;
  zlo[(size_t)row * DIM + d] = l;

  __shared__ float red[256];
  red[d] = v * v;
  __syncthreads();
  for (int s = 128; s > 0; s >>= 1) {
    if (d < s) red[d] += red[d + s];
    __syncthreads();
  }
  if (d == 0) z2[row] = red[0];
}

// ---------------------------------------------------------------------------
// Kernel 2: 128x128 Gram tiles via v_wmma_f32_16x16x32_f16 (hi/lo split),
// fused exp epilogue, per-block partial sum. Upper-triangular tiles only.
// 256 threads = 8 waves; wave w owns rows [w*16, w*16+16) of the tile.
// ---------------------------------------------------------------------------
__global__ __launch_bounds__(256) void mmd_tiles(
    const half_t* __restrict__ zhi, const half_t* __restrict__ zlo,
    const float* __restrict__ z2, float* __restrict__ partials)
{
  const int tc  = blockIdx.x;
  const int tr  = blockIdx.y;
  const int bid = tr * NT + tc;
  if (tc < tr) {                      // lower triangle: emit zero, exit
    if (threadIdx.x == 0) partials[bid] = 0.0f;
    return;
  }

  const int lane = threadIdx.x & 31;
  const int wave = threadIdx.x >> 5;
  const int m    = lane & 15;         // row-within-fragment / column-within-fragment
  const int hsel = lane >> 4;         // lane-half selector
  const int row0 = tr * TILE + wave * 16;
  const int col0 = tc * TILE;

  v8f acc[8] = {};

  const half_t* arow_h  = zhi + (size_t)(row0 + m) * DIM;
  const half_t* arow_l  = zlo + (size_t)(row0 + m) * DIM;
  const half_t* bbase_h = zhi + (size_t)(col0 + m) * DIM;
  const half_t* bbase_l = zlo + (size_t)(col0 + m) * DIM;

  for (int kb = 0; kb < DIM; kb += 32) {
    // A fragment (16x32 f16): lanes 0-15 K = kb+0..15 of row m;
    //                         lanes 16-31 K = kb+16..31.
    const int ka = kb + hsel * 16;
    const v8h ah0 = *(const v8h*)(arow_h + ka);
    const v8h ah1 = *(const v8h*)(arow_h + ka + 8);
    const v8h al0 = *(const v8h*)(arow_l + ka);
    const v8h al1 = *(const v8h*)(arow_l + ka + 8);
    const v16h a_hi = __builtin_shufflevector(ah0, ah1, 0,1,2,3,4,5,6,7,8,9,10,11,12,13,14,15);
    const v16h a_lo = __builtin_shufflevector(al0, al1, 0,1,2,3,4,5,6,7,8,9,10,11,12,13,14,15);

    // B fragment (32x16 f16): lane holds 16 consecutive K of one Z row.
    const int kbb = kb + hsel * 16;
#pragma unroll
    for (int t = 0; t < 8; ++t) {
      const v16h b_hi = *(const v16h*)(bbase_h + (size_t)t * 16 * DIM + kbb);
      const v16h b_lo = *(const v16h*)(bbase_l + (size_t)t * 16 * DIM + kbb);
      acc[t] = __builtin_amdgcn_wmma_f32_16x16x32_f16(false, a_hi, false, b_hi,
                                                      (short)0, acc[t], false, false);
      acc[t] = __builtin_amdgcn_wmma_f32_16x16x32_f16(false, a_hi, false, b_lo,
                                                      (short)0, acc[t], false, false);
      acc[t] = __builtin_amdgcn_wmma_f32_16x16x32_f16(false, a_lo, false, b_hi,
                                                      (short)0, acc[t], false, false);
    }
  }

  // Epilogue: sq = zi + zj - 2*dot, clamp, exp(-0.5*sq/ell2) via exp2.
  float zi[8];
#pragma unroll
  for (int r = 0; r < 8; ++r) zi[r] = z2[row0 + hsel * 8 + r];  // M = r + hsel*8

  // tile-uniform sign (8192 is a multiple of TILE) and triangle weight
  const float sgn = ((tr < NT / 2) == (tc < NT / 2)) ? 1.0f : -1.0f;
  const float wgt = (tc > tr) ? 2.0f * sgn : sgn;
  const float scale = -(1.0f / 256.0f) * 1.4426950408889634f;  // -0.5/ell2 * log2(e)

  float local = 0.0f;
#pragma unroll
  for (int t = 0; t < 8; ++t) {
    const float zj = z2[col0 + t * 16 + m];                    // N = lane&15
#pragma unroll
    for (int r = 0; r < 8; ++r) {
      float sq = zi[r] + zj - 2.0f * acc[t][r];
      sq = fmaxf(sq, 0.0f);
      local += exp2f(sq * scale);
    }
  }
  local *= wgt;

  __shared__ float red[256];
  red[threadIdx.x] = local;
  __syncthreads();
  for (int s = 128; s > 0; s >>= 1) {
    if (threadIdx.x < s) red[threadIdx.x] += red[threadIdx.x + s];
    __syncthreads();
  }
  if (threadIdx.x == 0) partials[bid] = red[0];
}

// ---------------------------------------------------------------------------
// Kernel 3: deterministic final reduction (double) + sqrt.
// ---------------------------------------------------------------------------
__global__ __launch_bounds__(256) void mmd_final(
    const float* __restrict__ partials, float* __restrict__ out)
{
  __shared__ double red[256];
  double s = 0.0;
  for (int i = threadIdx.x; i < NT * NT; i += 256) s += (double)partials[i];
  red[threadIdx.x] = s;
  __syncthreads();
  for (int st = 128; st > 0; st >>= 1) {
    if (threadIdx.x < st) red[threadIdx.x] += red[threadIdx.x + st];
    __syncthreads();
  }
  if (threadIdx.x == 0) {
    double mmd = red[0] / ((double)NX * (double)NX);
    if (mmd < 0.0) mmd = 0.0;
    out[0] = (float)sqrt(mmd);
  }
}

// ---------------------------------------------------------------------------
extern "C" void kernel_launch(void* const* d_in, const int* in_sizes, int n_in,
                              void* d_out, int out_size, void* d_ws, size_t ws_size,
                              hipStream_t stream)
{
  (void)in_sizes; (void)n_in; (void)out_size; (void)ws_size;
  const float* x = (const float*)d_in[0];
  const float* y = (const float*)d_in[1];
  float* out = (float*)d_out;

  char* ws = (char*)d_ws;
  const size_t zbytes = (size_t)ZROWS * DIM * sizeof(half_t);  // 8 MB each
  half_t* zhi = (half_t*)ws;
  half_t* zlo = (half_t*)(ws + zbytes);
  float*  z2  = (float*)(ws + 2 * zbytes);
  float*  partials = z2 + ZROWS;
  // total workspace: 2*8MB + 64KB + 64KB ~= 16.9 MB

  mmd_prep<<<ZROWS, 256, 0, stream>>>(x, y, zhi, zlo, z2);
  dim3 grid(NT, NT, 1);
  mmd_tiles<<<grid, 256, 0, stream>>>(zhi, zlo, z2, partials);
  mmd_final<<<1, 256, 0, stream>>>(partials, out);
}
